// MLP4_47914655154754
// MI455X (gfx1250) — matrix operand order repaired and checked by reading, not compiled, not run on hardware
//
#include <hip/hip_runtime.h>
#include <hip/hip_bf16.h>

typedef _Float16 half_t;
typedef __attribute__((ext_vector_type(16))) _Float16 v16h;
typedef __attribute__((ext_vector_type(8)))  _Float16 v8h;
typedef __attribute__((ext_vector_type(8)))  float    v8f;

#define NLAYERS 17
#define WAVES_PER_BLOCK 4
#define MROWS 32                  // samples per wave (2 WMMA M-tiles) -> 2x B reuse
#define ACT_STRIDE 328            // halfs per row; 164 dwords, gcd(164,64)=4 -> conflict-free
#define BUF_HALFS (MROWS * ACT_STRIDE)
#define WTOT 526848               // total padded f16 weight elements

#define CAT16(lo, hi) __builtin_shufflevector((lo), (hi), \
        0,1,2,3,4,5,6,7,8,9,10,11,12,13,14,15)

// ---- per-layer tables (compile-time constants of the reference net) ----
__constant__ int c_kt  [NLAYERS] = {1,1,2,2,5,5,5,10,10,10,10,5,5,5,2,2,1};       // ceil(fin/32)
__constant__ int c_nt  [NLAYERS] = {1,4,4,10,10,10,19,19,19,19,10,10,10,4,4,1,1}; // ceil(fout/16)
__constant__ int c_woff[NLAYERS] = {0,512,2560,6656,16896,42496,68096,116736,214016,
                                    311296,408576,459776,485376,510976,521216,525312,526336};
__constant__ int c_boff[NLAYERS] = {0,16,80,144,304,464,624,928,1232,1536,1840,
                                    2000,2160,2320,2384,2448,2464};
__constant__ int c_pad [NLAYERS] = {16,0,0,0,0,0,16,16,16,16,0,0,0,0,0,16,0};     // Kp_next - Np

// host-side copies
static const int h_fin [NLAYERS] = {5,15,50,50,150,150,150,300,300,300,300,150,150,150,50,50,15};
static const int h_fout[NLAYERS] = {15,50,50,150,150,150,300,300,300,300,150,150,150,50,50,15,10};
static const int h_kt  [NLAYERS] = {1,1,2,2,5,5,5,10,10,10,10,5,5,5,2,2,1};
static const int h_nt  [NLAYERS] = {1,4,4,10,10,10,19,19,19,19,10,10,10,4,4,1,1};
static const int h_woff[NLAYERS] = {0,512,2560,6656,16896,42496,68096,116736,214016,
                                    311296,408576,459776,485376,510976,521216,525312,526336};
static const int h_boff[NLAYERS] = {0,16,80,144,304,464,624,928,1232,1536,1840,
                                    2000,2160,2320,2384,2448,2464};

// ---------------------------------------------------------------------------
// Prep: f32 row-major W[fout][fin] -> WMMA B-tile f16 layout.
// ---------------------------------------------------------------------------
__global__ void prep_layer(const float* __restrict__ W, const float* __restrict__ B,
                           half_t* __restrict__ wdst, float* __restrict__ bdst,
                           int fin, int fout, int kt, int nt)
{
    int t = blockIdx.x * blockDim.x + threadIdx.x;
    int total = kt * nt * 512;
    if (t < total) {
        int e    = t & 15;
        int lane = (t >> 4) & 31;
        int tile = t >> 9;
        int ktl  = tile % kt;
        int ntl  = tile / kt;
        int K = ktl * 32 + ((e < 8) ? 0 : 8) + ((lane < 16) ? 0 : 8) + e;
        int N = ntl * 16 + (lane & 15);
        float v = (K < fin && N < fout) ? W[(size_t)N * fin + K] : 0.0f;
        wdst[t] = (half_t)v;
    }
    if (t < nt * 16) {
        bdst[t] = (t < fout) ? B[t] : 0.0f;
    }
}

// ---------------------------------------------------------------------------
// One layer with KT compile-time K-tiles: A fragments for both M-tiles are
// loaded from LDS ONCE into registers, then reused across all N-tiles while
// only B fragments stream from global/L2. 2 wmmas per B load.
// ---------------------------------------------------------------------------
template<int KT>
__device__ __forceinline__ void run_layer(const _Float16* __restrict__ cur,
                                          _Float16* __restrict__ nxt,
                                          const half_t* __restrict__ wl,
                                          const float* __restrict__ bl,
                                          int nt, bool last, int lane)
{
    const int arow  = lane & 15;
    const int koff0 = (lane < 16) ? 0 : 8;      // A-fragment lane K offset
    const int ncol  = lane & 15;                // C/D column within tile

    v16h a0[KT], a1[KT];                        // register-resident A fragments
    #pragma unroll
    for (int ktl = 0; ktl < KT; ++ktl) {
        const _Float16* ap0 = cur + arow * ACT_STRIDE + ktl * 32 + koff0;
        const _Float16* ap1 = ap0 + 16 * ACT_STRIDE;
        v8h lo0 = *(const v8h*)(ap0);
        v8h hi0 = *(const v8h*)(ap0 + 16);
        v8h lo1 = *(const v8h*)(ap1);
        v8h hi1 = *(const v8h*)(ap1 + 16);
        a0[ktl] = CAT16(lo0, hi0);
        a1[ktl] = CAT16(lo1, hi1);
    }

    for (int ntl = 0; ntl < nt; ++ntl) {
        const int N = ntl * 16 + ncol;
        const float bias = bl[N];               // zero-padded -> no guard
        const half_t* wt = wl + (size_t)(ntl * KT) * 512 + (size_t)lane * 16;
        v8f acc0 = {};
        v8f acc1 = {};
        #pragma unroll
        for (int ktl = 0; ktl < KT; ++ktl) {
            v16h b = *(const v16h*)(wt + (size_t)ktl * 512);
            acc0 = __builtin_amdgcn_wmma_f32_16x16x32_f16(
                    false, a0[ktl], false, b, (short)0, acc0, false, false);
            acc1 = __builtin_amdgcn_wmma_f32_16x16x32_f16(
                    false, a1[ktl], false, b, (short)0, acc1, false, false);
        }
        if (!last) {
            #pragma unroll
            for (int r = 0; r < 8; ++r) {
                int m = (lane < 16) ? r : r + 8;
                float v0 = acc0[r] + bias;
                v0 = fmaxf(v0, 0.0f) + 0.01f * fminf(v0, 0.0f);   // LeakyReLU
                nxt[m * ACT_STRIDE + N] = (_Float16)v0;
                float v1 = acc1[r] + bias;
                v1 = fmaxf(v1, 0.0f) + 0.01f * fminf(v1, 0.0f);
                nxt[(m + 16) * ACT_STRIDE + N] = (_Float16)v1;
            }
        } else {
            float* cb = (float*)nxt;            // final 32x16 f32 coefficients
            #pragma unroll
            for (int r = 0; r < 8; ++r) {
                int m = (lane < 16) ? r : r + 8;
                cb[m * 16 + ncol]        = acc0[r] + bias;
                cb[(m + 16) * 16 + ncol] = acc1[r] + bias;
            }
        }
    }
}

// ---------------------------------------------------------------------------
// Main: one wave = 32 samples (two 16-row M-tiles) through the whole MLP.
// ---------------------------------------------------------------------------
extern __shared__ _Float16 smem[];

__global__ void mlp_wmma_kernel(const float* __restrict__ x,
                                const float* __restrict__ y,
                                const half_t* __restrict__ wts,
                                const float* __restrict__ biases,
                                float* __restrict__ out, int nsamp)
{
    const int lane = threadIdx.x & 31;
    const int wid  = threadIdx.x >> 5;
    const long tile = (long)blockIdx.x * WAVES_PER_BLOCK + wid;
    const long s0 = tile * MROWS;
    if (s0 >= nsamp) return;                    // no barriers used -> safe exit

    _Float16* buf0 = smem + (size_t)wid * (2 * BUF_HALFS);
    _Float16* buf1 = buf0 + BUF_HALFS;

    // ---- stage x[32][5] into buffer 0, zero-padded to 32 columns ----
    {
        int cbase = (lane < 16) ? 0 : 16;
        #pragma unroll
        for (int mt = 0; mt < 2; ++mt) {
            int m = (lane & 15) + mt * 16;
            const float* xr = x + (s0 + m) * 5;
            #pragma unroll
            for (int c = 0; c < 16; ++c) {
                int col = cbase + c;
                float v = (col < 5) ? xr[col] : 0.0f;
                buf0[m * ACT_STRIDE + col] = (_Float16)v;
            }
        }
    }

    _Float16* cur = buf0;
    _Float16* nxt = buf1;

    for (int l = 0; l < NLAYERS; ++l) {
        const int nt = c_nt[l];
        const half_t* wl = wts + c_woff[l];
        const float*  bl = biases + c_boff[l];
        const bool last = (l == NLAYERS - 1);
        __builtin_prefetch(wl + (size_t)lane * 16, 0, 1);

        switch (c_kt[l]) {
            case 1:  run_layer<1> (cur, nxt, wl, bl, nt, last, lane); break;
            case 2:  run_layer<2> (cur, nxt, wl, bl, nt, last, lane); break;
            case 5:  run_layer<5> (cur, nxt, wl, bl, nt, last, lane); break;
            default: run_layer<10>(cur, nxt, wl, bl, nt, last, lane); break;
        }

        // zero K-padding columns needed by the next layer (Np < Kp_next)
        if (!last && c_pad[l]) {
            v8h z = {};
            _Float16* zp = nxt + lane * ACT_STRIDE + nt * 16;  // lane = row 0..31
            *(v8h*)(zp)     = z;
            *(v8h*)(zp + 8) = z;
        }
        _Float16* t = cur; cur = nxt; nxt = t;
    }

    // cross-lane LDS handoff of final coefficients -> make writes visible
    asm volatile("s_wait_dscnt 0" ::: "memory");

    // ---- einsum: out[n,k] = sum_b C[n,b] * y[n,b,k]  (b=10, k=9) ----
    {
        const float* cb = (const float*)cur;
        float c[10];
        #pragma unroll
        for (int b = 0; b < 10; ++b) c[b] = cb[lane * 16 + b];   // one sample/lane
        const float* yr = y + (s0 + lane) * 90;
        float* orow = out + (s0 + lane) * 9;
        #pragma unroll
        for (int k = 0; k < 9; ++k) {
            float s = 0.0f;
            #pragma unroll
            for (int b = 0; b < 10; ++b) s += c[b] * yr[b * 9 + k];
            orow[k] = s;
        }
    }
}

// ---------------------------------------------------------------------------
extern "C" void kernel_launch(void* const* d_in, const int* in_sizes, int n_in,
                              void* d_out, int out_size, void* d_ws, size_t ws_size,
                              hipStream_t stream)
{
    const float* x = (const float*)d_in[0];
    const float* y = (const float*)d_in[1];

    char*   wsb  = (char*)d_ws;
    half_t* wbuf = (half_t*)wsb;                          // WTOT f16 weights
    float*  bbuf = (float*)(wsb + (size_t)WTOT * 2);      // padded biases

    for (int l = 0; l < NLAYERS; ++l) {
        const float* W = (const float*)d_in[2 + 2 * l];
        const float* B = (const float*)d_in[3 + 2 * l];
        int total  = h_kt[l] * h_nt[l] * 512;
        int blocks = (total + 255) / 256;
        prep_layer<<<blocks, 256, 0, stream>>>(
            W, B, wbuf + h_woff[l], bbuf + h_boff[l],
            h_fin[l], h_fout[l], h_kt[l], h_nt[l]);
    }

    int  nsamp  = in_sizes[0] / 5;
    long tiles  = ((long)nsamp + MROWS - 1) / MROWS;
    int  blocks = (int)((tiles + WAVES_PER_BLOCK - 1) / WAVES_PER_BLOCK);
    size_t shmem = (size_t)WAVES_PER_BLOCK * 2 * BUF_HALFS * sizeof(_Float16); // 168 KB

    mlp_wmma_kernel<<<blocks, WAVES_PER_BLOCK * 32, shmem, stream>>>(
        x, y, wbuf, bbuf, (float*)d_out, nsamp);
}
